// RetentionUnit_55405078118791
// MI455X (gfx1250) — matrix operand order repaired and checked by reading, not compiled, not run on hardware
//
#include <hip/hip_runtime.h>

#define B_SZ 8
#define T_SZ 2048
#define D_SZ 1024
#define M_SZ (B_SZ * T_SZ)   // 16384 rows
#define NC 16                // scan chunks
#define CL (T_SZ / NC)       // 128 steps per chunk

typedef __attribute__((ext_vector_type(16))) __bf16 v16bf;
typedef __attribute__((ext_vector_type(8)))  float  v8f;

// ---------------------------------------------------------------- bf16 cvt
__device__ __forceinline__ unsigned short f2bf(float f) {
    unsigned int u = __float_as_uint(f);
    u += 0x7FFFu + ((u >> 16) & 1u);   // round-to-nearest-even
    return (unsigned short)(u >> 16);
}

__global__ __launch_bounds__(256) void cvt_x_kernel(const float* __restrict__ x,
                                                    unsigned short* __restrict__ xb,
                                                    int n4) {
    int i = blockIdx.x * blockDim.x + threadIdx.x;
    if (i < n4) {
        float4 f = ((const float4*)x)[i];
        ushort4 h;
        h.x = f2bf(f.x); h.y = f2bf(f.y); h.z = f2bf(f.z); h.w = f2bf(f.w);
        ((ushort4*)xb)[i] = h;
    }
}

__global__ __launch_bounds__(256) void cvt_w_kernel(const float* __restrict__ wq,
                                                    const float* __restrict__ wk,
                                                    const float* __restrict__ wv,
                                                    unsigned short* __restrict__ wb) {
    const int per = D_SZ * D_SZ / 4;
    int i = blockIdx.x * blockDim.x + threadIdx.x;   // 0 .. 3*per-1
    const float* src = (i < per) ? wq : ((i < 2 * per) ? wk : wv);
    int j = (i < per) ? i : ((i < 2 * per) ? (i - per) : (i - 2 * per));
    float4 f = ((const float4*)src)[j];
    ushort4 h;
    h.x = f2bf(f.x); h.y = f2bf(f.y); h.z = f2bf(f.z); h.w = f2bf(f.w);
    ((ushort4*)wb)[i] = h;
}

// ---------------------------------------------------------------- fragment loads
// A (16x32 bf16, MxK): lane lr holds row M=lr; lane-half lh selects K base lh*16,
// elements 0..15 -> K = lh*16 .. +15 (contiguous, matches B)
__device__ __forceinline__ v16bf load_frag_a(const unsigned short* __restrict__ p) {
    v16bf v;
    ((uint4*)&v)[0] = *(const uint4*)(p);
    ((uint4*)&v)[1] = *(const uint4*)(p + 8);
    return v;
}
// B (32x16 bf16, KxN): lane lr holds column N=lr; lanes 0-15 hold K=0..15,
// lanes 16-31 hold K=16..31 (contiguous per lane, same order as A)
__device__ __forceinline__ v16bf load_frag_b(const unsigned short* __restrict__ p) {
    v16bf v;
    ((uint4*)&v)[0] = *(const uint4*)(p);
    ((uint4*)&v)[1] = *(const uint4*)(p + 8);
    return v;
}

#define WMMA_BF16(A, Bm, C) \
    __builtin_amdgcn_wmma_f32_16x16x32_bf16(false, (A), false, (Bm), (short)0, (C), false, false)

// ---------------------------------------------------------------- fused q/k/v GEMM
// grid: (M/128, D/64), block 256 (8 waves). Wave -> 32x32 tile of each of q,k,v.
__global__ __launch_bounds__(256) void qkv_gemm_kernel(
        const unsigned short* __restrict__ xb,   // [M, D] bf16
        const unsigned short* __restrict__ wb,   // [3, D, D] bf16 (q,k,v), row = out-feature
        const float* __restrict__ bq, const float* __restrict__ bk,
        const float* __restrict__ bv,
        float* __restrict__ qout,                // [M, D]
        float* __restrict__ kvout) {             // [M, D]
    const int lane = threadIdx.x & 31;
    const int wave = threadIdx.x >> 5;
    const int lr = lane & 15;
    const int lh = lane >> 4;
    const int m0 = blockIdx.x * 128 + (wave & 3) * 32;
    const int n0 = blockIdx.y * 64  + (wave >> 2) * 32;

    v8f accq[2][2], acck[2][2], accv[2][2];
#pragma unroll
    for (int i = 0; i < 2; ++i)
#pragma unroll
        for (int j = 0; j < 2; ++j) {
            accq[i][j] = (v8f){0.f,0.f,0.f,0.f,0.f,0.f,0.f,0.f};
            acck[i][j] = (v8f){0.f,0.f,0.f,0.f,0.f,0.f,0.f,0.f};
            accv[i][j] = (v8f){0.f,0.f,0.f,0.f,0.f,0.f,0.f,0.f};
        }

    // per-lane base pointers
    const unsigned short* aP0 = xb + (size_t)(m0 + lr) * D_SZ + lh * 16;
    const unsigned short* aP1 = aP0 + (size_t)16 * D_SZ;

    const size_t wsz = (size_t)D_SZ * D_SZ;
    const size_t row0 = (size_t)(n0 + lr) * D_SZ + lh * 16;
    const size_t row1 = (size_t)(n0 + 16 + lr) * D_SZ + lh * 16;
    const unsigned short* bq0 = wb + row0;
    const unsigned short* bq1 = wb + row1;
    const unsigned short* bk0 = wb + wsz + row0;
    const unsigned short* bk1 = wb + wsz + row1;
    const unsigned short* bv0 = wb + 2 * wsz + row0;
    const unsigned short* bv1 = wb + 2 * wsz + row1;

    for (int kb = 0; kb < D_SZ; kb += 32) {
        v16bf A0  = load_frag_a(aP0 + kb);
        v16bf A1  = load_frag_a(aP1 + kb);
        v16bf Bq0 = load_frag_b(bq0 + kb);
        v16bf Bq1 = load_frag_b(bq1 + kb);
        v16bf Bk0 = load_frag_b(bk0 + kb);
        v16bf Bk1 = load_frag_b(bk1 + kb);
        v16bf Bv0 = load_frag_b(bv0 + kb);
        v16bf Bv1 = load_frag_b(bv1 + kb);

        accq[0][0] = WMMA_BF16(A0, Bq0, accq[0][0]);
        accq[0][1] = WMMA_BF16(A0, Bq1, accq[0][1]);
        accq[1][0] = WMMA_BF16(A1, Bq0, accq[1][0]);
        accq[1][1] = WMMA_BF16(A1, Bq1, accq[1][1]);
        acck[0][0] = WMMA_BF16(A0, Bk0, acck[0][0]);
        acck[0][1] = WMMA_BF16(A0, Bk1, acck[0][1]);
        acck[1][0] = WMMA_BF16(A1, Bk0, acck[1][0]);
        acck[1][1] = WMMA_BF16(A1, Bk1, acck[1][1]);
        accv[0][0] = WMMA_BF16(A0, Bv0, accv[0][0]);
        accv[0][1] = WMMA_BF16(A0, Bv1, accv[0][1]);
        accv[1][0] = WMMA_BF16(A1, Bv0, accv[1][0]);
        accv[1][1] = WMMA_BF16(A1, Bv1, accv[1][1]);
    }

    // epilogue: biases, kv = k*v, store. C layout: VGPR r -> (M = r + 8*lh, N = lr)
    float bqv[2], bkv[2], bvv[2];
    bqv[0] = bq[n0 + lr];      bqv[1] = bq[n0 + 16 + lr];
    bkv[0] = bk[n0 + lr];      bkv[1] = bk[n0 + 16 + lr];
    bvv[0] = bv[n0 + lr];      bvv[1] = bv[n0 + 16 + lr];

#pragma unroll
    for (int i = 0; i < 2; ++i) {
        const int mb = m0 + 16 * i + 8 * lh;
#pragma unroll
        for (int j = 0; j < 2; ++j) {
            const int n = n0 + 16 * j + lr;
#pragma unroll
            for (int r = 0; r < 8; ++r) {
                size_t o = (size_t)(mb + r) * D_SZ + n;
                float qv  = accq[i][j][r] + bqv[j];
                float kvv = (acck[i][j][r] + bkv[j]) * (accv[i][j][r] + bvv[j]);
                qout[o]  = qv;
                kvout[o] = kvv;
            }
        }
    }
}

// ---------------------------------------------------------------- chunked scan
// gid = b*NC*D + c*D + d   (adjacent threads -> adjacent d -> coalesced)
__global__ __launch_bounds__(256) void scan_partial_kernel(const float* __restrict__ kv,
                                                           const float* __restrict__ decay,
                                                           float* __restrict__ E) {
    int gid = blockIdx.x * blockDim.x + threadIdx.x;   // B*NC*D
    int d = gid & (D_SZ - 1);
    int c = (gid >> 10) & (NC - 1);
    int b = gid >> 14;
    float dec = decay[d];
    size_t idx = ((size_t)b * T_SZ + (size_t)c * CL) * D_SZ + d;
    float mem = 0.f;
#pragma unroll 4
    for (int t = 0; t < CL; ++t) {
        mem = dec * mem + kv[idx];
        idx += D_SZ;
    }
    E[gid] = mem;
}

__global__ __launch_bounds__(256) void scan_carry_kernel(const float* __restrict__ E,
                                                         const float* __restrict__ decay,
                                                         float* __restrict__ S) {
    int gid = blockIdx.x * blockDim.x + threadIdx.x;   // B*D
    int d = gid & (D_SZ - 1);
    int b = gid >> 10;
    float dec = decay[d];
    float dL = dec;
#pragma unroll
    for (int i = 0; i < 7; ++i) dL *= dL;              // dec^128 == dec^CL
    float s = 0.f;
    size_t base = (size_t)b * NC * D_SZ + d;
    for (int c = 0; c < NC; ++c) {
        S[base + (size_t)c * D_SZ] = s;
        s = dL * s + E[base + (size_t)c * D_SZ];
    }
}

__global__ __launch_bounds__(256) void scan_apply_kernel(const float* __restrict__ kv,
                                                         const float* __restrict__ decay,
                                                         const float* __restrict__ S,
                                                         float* __restrict__ out) {
    int gid = blockIdx.x * blockDim.x + threadIdx.x;   // B*NC*D
    int d = gid & (D_SZ - 1);
    int c = (gid >> 10) & (NC - 1);
    int b = gid >> 14;
    float dec = decay[d];
    float mem = S[gid];
    size_t idx = ((size_t)b * T_SZ + (size_t)c * CL) * D_SZ + d;
#pragma unroll 4
    for (int t = 0; t < CL; ++t) {
        mem = dec * mem + kv[idx];
        out[idx] = out[idx] * mem;    // out currently holds q
        idx += D_SZ;
    }
}

// ---------------------------------------------------------------- launch
extern "C" void kernel_launch(void* const* d_in, const int* in_sizes, int n_in,
                              void* d_out, int out_size, void* d_ws, size_t ws_size,
                              hipStream_t stream) {
    (void)in_sizes; (void)n_in; (void)out_size; (void)ws_size;
    const float* x     = (const float*)d_in[0];
    const float* Wq    = (const float*)d_in[1];
    const float* bq    = (const float*)d_in[2];
    const float* Wk    = (const float*)d_in[3];
    const float* bk    = (const float*)d_in[4];
    const float* Wv    = (const float*)d_in[5];
    const float* bv    = (const float*)d_in[6];
    const float* decay = (const float*)d_in[7];
    float* out = (float*)d_out;

    char* ws = (char*)d_ws;
    size_t off = 0;
    unsigned short* xb  = (unsigned short*)(ws + off); off += (size_t)M_SZ * D_SZ * 2;       // 32 MB
    unsigned short* wbf = (unsigned short*)(ws + off); off += (size_t)3 * D_SZ * D_SZ * 2;   //  6 MB
    float* kv = (float*)(ws + off); off += (size_t)M_SZ * D_SZ * 4;                          // 64 MB
    float* E  = (float*)(ws + off); off += (size_t)B_SZ * NC * D_SZ * 4;                     // 512 KB
    float* S  = (float*)(ws + off); off += (size_t)B_SZ * NC * D_SZ * 4;                     // 512 KB

    {
        int n4 = M_SZ * D_SZ / 4;
        cvt_x_kernel<<<n4 / 256, 256, 0, stream>>>(x, xb, n4);
    }
    cvt_w_kernel<<<(3 * D_SZ * D_SZ / 4) / 256, 256, 0, stream>>>(Wq, Wk, Wv, wbf);

    dim3 g(M_SZ / 128, D_SZ / 64);
    qkv_gemm_kernel<<<g, 256, 0, stream>>>(xb, wbf, bq, bk, bv, out, kv);

    scan_partial_kernel<<<(B_SZ * NC * D_SZ) / 256, 256, 0, stream>>>(kv, decay, E);
    scan_carry_kernel<<<(B_SZ * D_SZ) / 256, 256, 0, stream>>>(E, decay, S);
    scan_apply_kernel<<<(B_SZ * NC * D_SZ) / 256, 256, 0, stream>>>(kv, decay, S, out);
}
